// Encoder_34737695490645
// MI455X (gfx1250) — matrix operand order repaired and checked by reading, not compiled, not run on hardware
//
#include <hip/hip_runtime.h>
#include <hip/hip_bf16.h>
#include <math.h>

#define Bsz   8
#define Ssz   512
#define Esz   512
#define Hn    8
#define DVsz  512
#define Msz   4096
#define MIDsz 2048
#define Lnum  6
#define BSrows (Bsz * Ssz)   // 4096 token rows

typedef _Float16 v16h __attribute__((ext_vector_type(16)));
typedef _Float16 v8h  __attribute__((ext_vector_type(8)));
typedef float    v8f  __attribute__((ext_vector_type(8)));

union Frag16 { v16h v; v8h h2[2]; };
union Half4  { _Float16 h[4]; unsigned long long u64; };

// ---------------------------------------------------------------------------
// WMMA GEMM:  C[z] = alpha * A[z] * B[z] (+bias) (optional relu)
// Block = 256 threads = 8 waves. Block tile 128(M) x 128(N), K chunk 32.
// Wave (wm 0..3, wn 0..1) owns a 32x64 sub-tile: 2 A-frags x 4 B-frags ->
// 8 x v_wmma_f32_16x16x32_f16 per K step, fed by 12 x ds_load_b128.
// Double-buffered LDS + register prefetch: global b128 loads for tile k+1
// issue before the WMMAs of tile k; single barrier per K step.
// Per-batch base offsets: off = (z/HH)*o?b + (z%HH)*o?h  (handles [B,H] slicing).
// transB: B stored row-major [N x K] (used for Q*K^T).
// ---------------------------------------------------------------------------
__global__ __launch_bounds__(256)
void gemm_wmma_f16(const float* __restrict__ A, const float* __restrict__ Bg,
                   const float* __restrict__ bias, float* __restrict__ C,
                   int Kd, int lda, int ldb, int ldc,
                   int transB, float alpha, int relu, int HH,
                   long long oAb, long long oAh, long long oBb, long long oBh,
                   long long oCb, long long oCh)
{
    __shared__ __align__(16) _Float16 As [2][128][40];  // [buf][m][k]
    __shared__ __align__(16) _Float16 Bst[2][128][40];  // [buf][n][k]

    const int z  = blockIdx.z;
    const int zb = z / HH, zh = z % HH;
    A  += zb * oAb + zh * oAh;
    Bg += zb * oBb + zh * oBh;
    C  += zb * oCb + zh * oCh;

    const int tid  = threadIdx.x;
    const int lane = tid & 31;
    const int wid  = tid >> 5;
    const int wm   = wid & 3;        // 0..3  -> 32-row strip
    const int wn   = wid >> 2;       // 0..1  -> 64-col strip
    const int lm   = lane & 15;
    const int lh   = lane >> 4;

    const int m0 = blockIdx.y * 128;
    const int n0 = blockIdx.x * 128;

    // staging index precompute (quad = float4 granularity)
    const int q_row8[4] = { (tid + 0)   >> 3, (tid + 256) >> 3,
                            (tid + 512) >> 3, (tid + 768) >> 3 };   // /8 -> row
    const int q_kq   = tid & 7;                // k-quad within row (k = 4*q_kq)
    const int q_kk[4] = { (tid + 0) >> 5, (tid + 256) >> 5,
                          (tid + 512) >> 5, (tid + 768) >> 5 };     // NN: k row
    const int q_nq   = tid & 31;               // NN: n-quad (n = 4*q_nq)

    float4 areg[4], breg[4];

    auto load_tile = [&](int kb) {
        #pragma unroll
        for (int i = 0; i < 4; ++i)
            areg[i] = *(const float4*)&A[(long long)(m0 + q_row8[i]) * lda + kb + 4 * q_kq];
        if (transB) {
            #pragma unroll
            for (int i = 0; i < 4; ++i)
                breg[i] = *(const float4*)&Bg[(long long)(n0 + q_row8[i]) * ldb + kb + 4 * q_kq];
        } else {
            #pragma unroll
            for (int i = 0; i < 4; ++i)
                breg[i] = *(const float4*)&Bg[(long long)(kb + q_kk[i]) * ldb + n0 + 4 * q_nq];
        }
    };
    auto store_tile = [&](int buf) {
        #pragma unroll
        for (int i = 0; i < 4; ++i) {
            Half4 t;
            t.h[0] = (_Float16)areg[i].x; t.h[1] = (_Float16)areg[i].y;
            t.h[2] = (_Float16)areg[i].z; t.h[3] = (_Float16)areg[i].w;
            *(unsigned long long*)&As[buf][q_row8[i]][4 * q_kq] = t.u64;
        }
        if (transB) {
            #pragma unroll
            for (int i = 0; i < 4; ++i) {
                Half4 t;
                t.h[0] = (_Float16)breg[i].x; t.h[1] = (_Float16)breg[i].y;
                t.h[2] = (_Float16)breg[i].z; t.h[3] = (_Float16)breg[i].w;
                *(unsigned long long*)&Bst[buf][q_row8[i]][4 * q_kq] = t.u64;
            }
        } else {
            #pragma unroll
            for (int i = 0; i < 4; ++i) {
                Bst[buf][4 * q_nq + 0][q_kk[i]] = (_Float16)breg[i].x;
                Bst[buf][4 * q_nq + 1][q_kk[i]] = (_Float16)breg[i].y;
                Bst[buf][4 * q_nq + 2][q_kk[i]] = (_Float16)breg[i].z;
                Bst[buf][4 * q_nq + 3][q_kk[i]] = (_Float16)breg[i].w;
            }
        }
    };

    v8f acc[2][4] = {};

    load_tile(0);
    store_tile(0);
    __syncthreads();

    for (int kb = 0; kb < Kd; kb += 32) {
        const int cur = (kb >> 5) & 1;
        const bool more = (kb + 32) < Kd;
        if (more) load_tile(kb + 32);          // issue global b128 loads early

        // fragments (documented wave32 layouts):
        //  A 16x32: lane(h,m), elem j -> k = (j>=8?16:0) + 8h + (j&7)
        //  B 32x16: lane(h,n), elem j -> k = (j>=8?16:0) + 8h + (j&7)
        Frag16 af[2], bf[4];
        #pragma unroll
        for (int i = 0; i < 2; ++i) {
            const _Float16* ap = &As[cur][32 * wm + 16 * i + lm][8 * lh];
            af[i].h2[0] = *(const v8h*)ap;
            af[i].h2[1] = *(const v8h*)(ap + 16);
        }
        #pragma unroll
        for (int i = 0; i < 4; ++i) {
            const _Float16* bp = &Bst[cur][64 * wn + 16 * i + lm][8 * lh];
            bf[i].h2[0] = *(const v8h*)bp;
            bf[i].h2[1] = *(const v8h*)(bp + 16);
        }
        #pragma unroll
        for (int ai = 0; ai < 2; ++ai)
            #pragma unroll
            for (int bi = 0; bi < 4; ++bi)
                acc[ai][bi] = __builtin_amdgcn_wmma_f32_16x16x32_f16(
                    false, af[ai].v, false, bf[bi].v, (short)0, acc[ai][bi],
                    false, false);

        if (more) store_tile(cur ^ 1);         // fill the other buffer
        __syncthreads();
    }

    // C/D 16x16 f32: lane(h,n), VGPR r -> row = 8h + r, col = n
    #pragma unroll
    for (int ai = 0; ai < 2; ++ai) {
        const int row0 = m0 + 32 * wm + 16 * ai + 8 * lh;
        #pragma unroll
        for (int bi = 0; bi < 4; ++bi) {
            const int col = n0 + 64 * wn + 16 * bi + lm;
            const float bs = bias ? bias[col] : 0.f;
            #pragma unroll
            for (int r = 0; r < 8; ++r) {
                float v = acc[ai][bi][r] * alpha + bs;
                if (relu) v = fmaxf(v, 0.f);
                C[(long long)(row0 + r) * ldc + col] = v;
            }
        }
    }
}

// ---------------------------------------------------------------------------
// Row softmax over S=512 columns with boolean mask (True -> -1e10).
// ---------------------------------------------------------------------------
__global__ __launch_bounds__(256)
void softmax_mask(float* __restrict__ sc, const unsigned char* __restrict__ mask)
{
    __shared__ float red[256];
    const int gid = blockIdx.x;          // 0 .. B*H*S-1
    const int q   = gid % Ssz;
    const int z   = gid / Ssz;           // b*H + h
    const int b   = z / Hn;
    float* row = sc + (long long)z * Ssz * Ssz + (long long)q * Ssz;
    const unsigned char* mrow = mask + (long long)b * Ssz * Ssz + (long long)q * Ssz;

    const int t = threadIdx.x;
    float x0 = mrow[t]       ? -1e10f : row[t];
    float x1 = mrow[t + 256] ? -1e10f : row[t + 256];

    red[t] = fmaxf(x0, x1);
    __syncthreads();
    for (int s = 128; s > 0; s >>= 1) { if (t < s) red[t] = fmaxf(red[t], red[t + s]); __syncthreads(); }
    const float mx = red[0];
    __syncthreads();

    float e0 = expf(x0 - mx), e1 = expf(x1 - mx);
    red[t] = e0 + e1;
    __syncthreads();
    for (int s = 128; s > 0; s >>= 1) { if (t < s) red[t] += red[t + s]; __syncthreads(); }
    const float inv = 1.f / red[0];
    row[t]       = e0 * inv;
    row[t + 256] = e1 * inv;
}

// ---------------------------------------------------------------------------
// out = LayerNorm(a + b) * g + be   over E=512; one block per token row.
// ---------------------------------------------------------------------------
__global__ __launch_bounds__(256)
void add_layernorm(const float* __restrict__ a, const float* __restrict__ bb,
                   const float* __restrict__ g, const float* __restrict__ be,
                   float* __restrict__ out)
{
    __shared__ float red[256];
    const long long row = blockIdx.x;
    const float* ar = a  + row * Esz;
    const float* br = bb + row * Esz;
    const int t = threadIdx.x;

    float x0 = ar[t] + br[t];
    float x1 = ar[t + 256] + br[t + 256];

    red[t] = x0 + x1;
    __syncthreads();
    for (int s = 128; s > 0; s >>= 1) { if (t < s) red[t] += red[t + s]; __syncthreads(); }
    const float mu = red[0] * (1.f / Esz);
    __syncthreads();

    const float d0 = x0 - mu, d1 = x1 - mu;
    red[t] = d0 * d0 + d1 * d1;
    __syncthreads();
    for (int s = 128; s > 0; s >>= 1) { if (t < s) red[t] += red[t + s]; __syncthreads(); }
    const float rstd = rsqrtf(red[0] * (1.f / Esz) + 1e-5f);

    out[row * Esz + t]       = d0 * rstd * g[t] + be[t];
    out[row * Esz + t + 256] = d1 * rstd * g[t + 256] + be[t + 256];
}

// ---------------------------------------------------------------------------
static inline void launch_gemm(hipStream_t st,
                               const float* A, const float* Bg, const float* bias, float* C,
                               int Mr, int Nc, int Kd, int lda, int ldb, int ldc,
                               int transB, float alpha, int relu,
                               int batches, int HH,
                               long long oAb, long long oAh,
                               long long oBb, long long oBh,
                               long long oCb, long long oCh)
{
    dim3 grid(Nc / 128, Mr / 128, batches);
    hipLaunchKernelGGL(gemm_wmma_f16, grid, dim3(256), 0, st,
                       A, Bg, bias, C, Kd, lda, ldb, ldc,
                       transB, alpha, relu, HH,
                       oAb, oAh, oBb, oBh, oCb, oCh);
}

extern "C" void kernel_launch(void* const* d_in, const int* in_sizes, int n_in,
                              void* d_out, int out_size, void* d_ws, size_t ws_size,
                              hipStream_t stream)
{
    const float*         src  = (const float*)d_in[0];
    const unsigned char* mask = (const unsigned char*)d_in[1];
    const float* Wq = (const float*)d_in[2];
    const float* bq = (const float*)d_in[3];
    const float* Wk = (const float*)d_in[4];
    const float* bk = (const float*)d_in[5];
    const float* Wv = (const float*)d_in[6];
    const float* bv = (const float*)d_in[7];
    const float* Wo = (const float*)d_in[8];
    const float* bo = (const float*)d_in[9];
    const float* W1 = (const float*)d_in[10];
    const float* b1 = (const float*)d_in[11];
    const float* W2 = (const float*)d_in[12];
    const float* b2 = (const float*)d_in[13];
    const float* g1 = (const float*)d_in[14];
    const float* be1= (const float*)d_in[15];
    const float* g2 = (const float*)d_in[16];
    const float* be2= (const float*)d_in[17];

    char* ws = (char*)d_ws;
    float* qb  = (float*)(ws + ((size_t)0   << 20));   //  64 MB  B*S*M
    float* kb  = (float*)(ws + ((size_t)64  << 20));   //  64 MB
    float* vb  = (float*)(ws + ((size_t)128 << 20));   //  64 MB
    float* sc  = (float*)(ws + ((size_t)192 << 20));   //  64 MB  B*H*S*S
    float* ao  = (float*)(ws + ((size_t)256 << 20));   //   8 MB  B*S*E
    float* x1  = (float*)(ws + ((size_t)264 << 20));   //   8 MB
    float* hm  = (float*)(ws + ((size_t)272 << 20));   //  32 MB  B*S*MID
    float* ffb = (float*)(ws + ((size_t)304 << 20));   //   8 MB
    float* xb  = (float*)(ws + ((size_t)312 << 20));   //   8 MB  next-layer x
    float* ctx = qb;   // reuse: q consumed once scores are built

    const float alpha_s = 1.0f / sqrtf((float)DVsz);
    const long long SM  = (long long)Ssz * Msz;    // per-batch stride in q/k/v/ctx
    const long long SS  = (long long)Ssz * Ssz;    // per-head stride in scores

    for (int l = 0; l < Lnum; ++l) {
        const float* x = (l == 0) ? src : xb;
        const float* Wq_l = Wq + (size_t)l * Esz * Msz;
        const float* Wk_l = Wk + (size_t)l * Esz * Msz;
        const float* Wv_l = Wv + (size_t)l * Esz * Msz;
        const float* Wo_l = Wo + (size_t)l * Msz * Esz;
        const float* W1_l = W1 + (size_t)l * Esz * MIDsz;
        const float* W2_l = W2 + (size_t)l * MIDsz * Esz;
        const float* bq_l = bq + (size_t)l * Msz;
        const float* bk_l = bk + (size_t)l * Msz;
        const float* bv_l = bv + (size_t)l * Msz;
        const float* bo_l = bo + (size_t)l * Esz;
        const float* b1_l = b1 + (size_t)l * MIDsz;
        const float* b2_l = b2 + (size_t)l * Esz;
        const float* g1_l = g1 + (size_t)l * Esz;
        const float* be1_l= be1+ (size_t)l * Esz;
        const float* g2_l = g2 + (size_t)l * Esz;
        const float* be2_l= be2+ (size_t)l * Esz;

        // Q/K/V projections: [4096 x 512] * [512 x 4096]
        launch_gemm(stream, x, Wq_l, bq_l, qb, BSrows, Msz, Esz, Esz, Msz, Msz,
                    0, 1.f, 0, 1, 1, 0,0, 0,0, 0,0);
        launch_gemm(stream, x, Wk_l, bk_l, kb, BSrows, Msz, Esz, Esz, Msz, Msz,
                    0, 1.f, 0, 1, 1, 0,0, 0,0, 0,0);
        launch_gemm(stream, x, Wv_l, bv_l, vb, BSrows, Msz, Esz, Esz, Msz, Msz,
                    0, 1.f, 0, 1, 1, 0,0, 0,0, 0,0);

        // scores = (Q * K^T) / sqrt(DV), batched over B*H (NT GEMM)
        launch_gemm(stream, qb, kb, nullptr, sc, Ssz, Ssz, DVsz, Msz, Msz, Ssz,
                    1, alpha_s, 0, Bsz * Hn, Hn,
                    SM, DVsz, SM, DVsz, (long long)Hn * SS, SS);

        // masked softmax over key dimension
        hipLaunchKernelGGL(softmax_mask, dim3(Bsz * Hn * Ssz), dim3(256), 0, stream, sc, mask);

        // ctx = P * V, batched over B*H (NN GEMM), written back in [B,S,M] layout
        launch_gemm(stream, sc, vb, nullptr, ctx, Ssz, DVsz, Ssz, Ssz, Msz, Msz,
                    0, 1.f, 0, Bsz * Hn, Hn,
                    (long long)Hn * SS, SS, SM, DVsz, SM, DVsz);

        // attn_out = ctx * Wo + bo
        launch_gemm(stream, ctx, Wo_l, bo_l, ao, BSrows, Esz, Msz, Msz, Esz, Esz,
                    0, 1.f, 0, 1, 1, 0,0, 0,0, 0,0);

        // x1 = LN(x + attn_out)
        hipLaunchKernelGGL(add_layernorm, dim3(BSrows), dim3(256), 0, stream,
                           x, ao, g1_l, be1_l, x1);

        // h = relu(x1 * W1 + b1)
        launch_gemm(stream, x1, W1_l, b1_l, hm, BSrows, MIDsz, Esz, Esz, MIDsz, MIDsz,
                    0, 1.f, 1, 1, 1, 0,0, 0,0, 0,0);

        // ff = h * W2 + b2
        launch_gemm(stream, hm, W2_l, b2_l, ffb, BSrows, Esz, MIDsz, MIDsz, Esz, Esz,
                    0, 1.f, 0, 1, 1, 0,0, 0,0, 0,0);

        // x2 = LN(x1 + ff) -> next-layer input (or final output)
        float* xo = (l == Lnum - 1) ? (float*)d_out : xb;
        hipLaunchKernelGGL(add_layernorm, dim3(BSrows), dim3(256), 0, stream,
                           x1, ffb, g2_l, be2_l, xo);
    }
}
